// DFEM_42228118454673
// MI455X (gfx1250) — compile-verified, matrix operand order checked by reading, not run on hardware
//
#include <hip/hip_runtime.h>
#include <hip/hip_bf16.h>

typedef __bf16 bf16_t;
typedef __attribute__((ext_vector_type(8)))  __bf16 v8bf;
typedef __attribute__((ext_vector_type(16))) __bf16 v16bf;
typedef __attribute__((ext_vector_type(8)))  float  v8f;

#define Hc 256
#define Wc 256
#define HWc 65536
#define Cc 64
#define Bc 2
#define KOFF 576    // offset-conv contraction depth (C*9)
#define KTOT 1728   // 3 branches * C*9
#define BPAD 72     // padded bf16 row length for the LDS B tile

static __device__ __forceinline__ v16bf mk16(v8bf lo, v8bf hi) {
  v16bf r;
  #pragma unroll
  for (int q = 0; q < 8; ++q) { r[q] = lo[q]; r[8 + q] = hi[q]; }
  return r;
}

// ---------------------------------------------------------------------------
// Setup A: pack wp_x/wp_y into a zero-padded 64 x 576 bf16 WMMA weight matrix
// (rows 0..17 = wp_x channels, 18..35 = wp_y channels, 36..63 = zero)
// ---------------------------------------------------------------------------
__global__ __launch_bounds__(256) void pack_offw(
    const float* __restrict__ wp_x, const float* __restrict__ wp_y,
    bf16_t* __restrict__ Aoff) {
  int idx = blockIdx.x * 256 + threadIdx.x;
  if (idx >= Cc * KOFF) return;
  int r = idx / KOFF;
  int k = idx % KOFF;
  float v = 0.f;
  if (r < 36) {
    const float* wp = (r < 18) ? wp_x : wp_y;
    int ch = (r < 18) ? r : r - 18;
    v = wp[ch * KOFF + k];
  }
  Aoff[idx] = (bf16_t)v;
}

// ---------------------------------------------------------------------------
// Setup B: fold the 1x1 fusion conv into the per-branch contraction weights.
// A[o][k] (64 x 1728), k = branch*576 + c*9 + tap, stored bf16 row-major.
// ---------------------------------------------------------------------------
__global__ __launch_bounds__(256) void fuse_weights(
    const float* __restrict__ wc_x, const float* __restrict__ wm,
    const float* __restrict__ wc_y, const float* __restrict__ wo,
    bf16_t* __restrict__ A) {
  int idx = blockIdx.x * 256 + threadIdx.x;
  if (idx >= Cc * KTOT) return;
  int o  = idx / KTOT;
  int k  = idx % KTOT;
  int br = k / 576;
  int kk = k % 576;
  const float* wsrc = (br == 0) ? wc_x : (br == 1 ? wm : wc_y);
  const float* worow = wo + o * (3 * Cc) + br * Cc;
  float s = 0.f;
  for (int j = 0; j < Cc; ++j) s += worow[j] * wsrc[j * 576 + kk];
  A[o * KTOT + k] = (bf16_t)s;
}

// ---------------------------------------------------------------------------
// Setup C: output bias (bo + wo_m*bm) and zero-padded 64-entry offset bias
// ---------------------------------------------------------------------------
__global__ void fuse_bias(const float* __restrict__ wo,
                          const float* __restrict__ bm,
                          const float* __restrict__ bo,
                          const float* __restrict__ bp_x,
                          const float* __restrict__ bp_y,
                          float* __restrict__ bias2,
                          float* __restrict__ bpoff) {
  int t = threadIdx.x;
  if (t < Cc) {
    float s = bo[t];
    for (int j = 0; j < Cc; ++j) s += wo[t * (3 * Cc) + Cc + j] * bm[j];
    bias2[t] = s;
  } else if (t < 2 * Cc) {
    int r = t - Cc;
    float v = 0.f;
    if (r < 18)      v = bp_x[r];
    else if (r < 36) v = bp_y[r - 18];
    bpoff[r] = v;
  }
}

// ---------------------------------------------------------------------------
// Fully fused kernel: per 32-pixel segment
//   1) offset conv via WMMA prologue (M=36 padded to 64, K=576) -> LDS
//   2) bilinear gather table from LDS offsets (branch-free: w=0/idx=0 pad)
//   3) main WMMA GEMM (M=64, K=1728) with gathered bf16 B tiles
// 8 waves = 4 oc-tiles x 2 pixel-tiles; K-chunk 64 (2 indep WMMAs/iter),
// ping-pong B buffers, 1 barrier/iter, dual accumulators (no D->D NOPs).
// ---------------------------------------------------------------------------
__global__ __launch_bounds__(256) void dfem_main(
    const float* __restrict__ x,
    const bf16_t* __restrict__ Aoff,
    const bf16_t* __restrict__ A,
    const float* __restrict__ bias2,
    const float* __restrict__ bpoff,
    float* __restrict__ out) {
  __shared__ int4   s_qidx[3 * 9 * 32];   // [br3][tap][px] -> 4 gather idx
  __shared__ float4 s_qw  [3 * 9 * 32];   // matching 4 bilinear weights
  __shared__ bf16_t s_B[2][32][BPAD];     // ping-pong [pixel][k-in-chunk]
  __shared__ float  s_off[64][32];        // offset rows (36 used, rest junk)

  const int tid = threadIdx.x;
  const int blk = blockIdx.x;
  const int tw = blk & 7;            // which 32-px segment in the row
  const int h  = (blk >> 3) & 255;
  const int b  = blk >> 11;
  const int w0 = tw * 32;

  const int lane  = tid & 31;
  const int wave  = tid >> 5;
  const int mtile = wave & 3;
  const int ntile = wave >> 2;
  const int hi    = lane >> 4;
  const int ln    = lane & 15;
  const int oc    = mtile * 16 + ln;
  const int pxr   = ntile * 16 + ln;

  const int kloc = tid & 63;         // fixed k within every 64-wide chunk
  const int px0  = tid >> 6;         // px = px0 + 4*i

  const float* xb = x + (size_t)b * Cc * HWc;

  // ================= PROLOGUE: offset conv on the matrix pipe =============
  {
    int tap = kloc % 9;
    int c   = kloc / 9;
    v8f oa0 = {}, oa1 = {};
    const bf16_t* aorow = Aoff + (size_t)oc * KOFF;
    for (int kc = 0; kc < KOFF / 64; ++kc) {
      bf16_t* __restrict__ Bb = &s_B[kc & 1][0][0];
      int r3 = tap / 3, c3 = tap - r3 * 3;
      int hh = h + r3 - 1;
      int hhc = min(max(hh, 0), Hc - 1);
      float mrow = ((unsigned)hh < (unsigned)Hc) ? 1.f : 0.f;
      const float* xrow = xb + c * HWc + hhc * Wc;
      #pragma unroll
      for (int i = 0; i < 8; ++i) {
        int px = px0 + i * 4;
        int ww = w0 + px + c3 - 1;
        int wwc = min(max(ww, 0), Wc - 1);
        float m = ((unsigned)ww < (unsigned)Wc) ? mrow : 0.f;
        Bb[px * BPAD + kloc] = (bf16_t)(m * xrow[wwc]);
      }
      tap += 1; c += 7;
      if (tap >= 9) { tap -= 9; c += 1; }
      __syncthreads();
      const bf16_t* ak = aorow + kc * 64;
      const bf16_t* brow = Bb + pxr * BPAD;
      v16bf af0 = mk16(*(const v8bf*)(ak + 8 * hi),
                       *(const v8bf*)(ak + 16 + 8 * hi));
      v16bf bf0 = mk16(*(const v8bf*)(brow + 8 * hi),
                       *(const v8bf*)(brow + 16 + 8 * hi));
      v16bf af1 = mk16(*(const v8bf*)(ak + 32 + 8 * hi),
                       *(const v8bf*)(ak + 48 + 8 * hi));
      v16bf bf1 = mk16(*(const v8bf*)(brow + 32 + 8 * hi),
                       *(const v8bf*)(brow + 48 + 8 * hi));
      oa0 = __builtin_amdgcn_wmma_f32_16x16x32_bf16(false, af0, false, bf0,
                                                    (short)0, oa0, false, false);
      oa1 = __builtin_amdgcn_wmma_f32_16x16x32_bf16(false, af1, false, bf1,
                                                    (short)0, oa1, false, false);
    }
    // branch-free: rows >= 36 write junk into padding rows, never read
    #pragma unroll
    for (int r = 0; r < 8; ++r) {
      int row = mtile * 16 + r + 8 * hi;
      s_off[row][pxr] = oa0[r] + oa1[r] + bpoff[row];
    }
  }
  __syncthreads();

  // ============ bilinear gather table for all 3 K-space branches ==========
  for (int e = tid; e < 3 * 9 * 32; e += 256) {
    int br3 = e / 288;               // 0=deform-x, 1=plain, 2=deform-y
    int rem = e % 288;
    int tap = rem >> 5;
    int px  = rem & 31;
    int w = w0 + px;
    int4 qi;  float4 qw;
    if (br3 == 1) {                  // plain 3x3 conv: single integer tap
      int hh = h + tap / 3 - 1;
      int ww = w + tap % 3 - 1;
      bool v = ((unsigned)hh < (unsigned)Hc) && ((unsigned)ww < (unsigned)Wc);
      qi = make_int4(v ? (hh * Wc + ww) : 0, 0, 0, 0);
      qw = make_float4(v ? 1.f : 0.f, 0.f, 0.f, 0.f);
    } else {
      int obr = (br3 == 0) ? 0 : 1;  // offset row block
      float ox = s_off[obr * 18 + tap][px];
      float oy = s_off[obr * 18 + 9 + tap][px];
      float pxf = ox + (float)(tap / 3 - 1) + (float)(h + 1);
      float pyf = oy + (float)(tap % 3 - 1) + (float)(w + 1);
      float fx = floorf(pxf), fy = floorf(pyf);
      const float hpm = (float)(Hc + 1), wpm = (float)(Wc + 1);
      float qltx = fminf(fmaxf(fx, 0.f), hpm);
      float qlty = fminf(fmaxf(fy, 0.f), wpm);
      float qrbx = fminf(fmaxf(fx + 1.f, 0.f), hpm);
      float qrby = fminf(fmaxf(fy + 1.f, 0.f), wpm);
      float pxc = fminf(fmaxf(pxf, 0.f), hpm);
      float pyc = fminf(fmaxf(pyf, 0.f), wpm);
      float wlx = 1.f + (qltx - pxc);
      float wrx = 1.f - (qrbx - pxc);
      float wly = 1.f + (qlty - pyc);
      float wry = 1.f - (qrby - pyc);
      int ilx = (int)qltx - 1, ily = (int)qlty - 1;  // padded -> unpadded
      int irx = (int)qrbx - 1, iry = (int)qrby - 1;
      bool vlx = (unsigned)ilx < (unsigned)Hc, vly = (unsigned)ily < (unsigned)Wc;
      bool vrx = (unsigned)irx < (unsigned)Hc, vry = (unsigned)iry < (unsigned)Wc;
      qi.x = (vlx && vly) ? (ilx * Wc + ily) : 0;
      qi.y = (vrx && vry) ? (irx * Wc + iry) : 0;
      qi.z = (vlx && vry) ? (ilx * Wc + iry) : 0;
      qi.w = (vrx && vly) ? (irx * Wc + ily) : 0;
      qw.x = (vlx && vly) ? wlx * wly : 0.f;
      qw.y = (vrx && vry) ? wrx * wry : 0.f;
      qw.z = (vlx && vry) ? wlx * wry : 0.f;
      qw.w = (vrx && vly) ? wrx * wly : 0.f;
    }
    s_qidx[e] = qi;
    s_qw[e]   = qw;
  }
  __syncthreads();

  // ======================= MAIN GEMM (M=64, K=1728) =======================
  int tap = kloc % 9;
  int c   = kloc / 9;
  int br3 = 0;
  v8f acc0 = {}, acc1 = {};
  const bf16_t* arow = A + (size_t)oc * KTOT;

  for (int kc = 0; kc < KTOT / 64; ++kc) {
    bf16_t* __restrict__ Bb = &s_B[kc & 1][0][0];
    {
      const float* xc = xb + c * HWc;
      const int sqb = (br3 * 9 + tap) * 32;
      #pragma unroll
      for (int i = 0; i < 8; ++i) {
        int px = px0 + i * 4;
        int4   qi = s_qidx[sqb + px];
        float4 qw = s_qw[sqb + px];
        float val = qw.x * xc[qi.x];
        val = fmaf(qw.y, xc[qi.y], val);
        val = fmaf(qw.z, xc[qi.z], val);
        val = fmaf(qw.w, xc[qi.w], val);
        Bb[px * BPAD + kloc] = (bf16_t)val;
      }
      tap += 1; c += 7;                    // advance k by 64 (= 7*9 + 1)
      if (tap >= 9) { tap -= 9; c += 1; }
      if (c >= Cc)  { c -= Cc; br3 += 1; }
    }
    __syncthreads();

    const bf16_t* ak = arow + kc * 64;
    __builtin_prefetch(ak + 64, 0, 1);     // next chunk of A
    const bf16_t* brow = Bb + pxr * BPAD;
    v16bf af0 = mk16(*(const v8bf*)(ak + 8 * hi),
                     *(const v8bf*)(ak + 16 + 8 * hi));
    v16bf bf0 = mk16(*(const v8bf*)(brow + 8 * hi),
                     *(const v8bf*)(brow + 16 + 8 * hi));
    v16bf af1 = mk16(*(const v8bf*)(ak + 32 + 8 * hi),
                     *(const v8bf*)(ak + 48 + 8 * hi));
    v16bf bf1 = mk16(*(const v8bf*)(brow + 32 + 8 * hi),
                     *(const v8bf*)(brow + 48 + 8 * hi));
    acc0 = __builtin_amdgcn_wmma_f32_16x16x32_bf16(false, af0, false, bf0,
                                                   (short)0, acc0, false, false);
    acc1 = __builtin_amdgcn_wmma_f32_16x16x32_bf16(false, af1, false, bf1,
                                                   (short)0, acc1, false, false);
  }

  // ---- epilogue: fused bias + residual
  const int wgl = w0 + pxr;
  #pragma unroll
  for (int r = 0; r < 8; ++r) {
    int ocr = mtile * 16 + r + 8 * hi;    // C/D layout: VGPR r -> M = r + 8*hi
    size_t o = ((size_t)(b * Cc + ocr) * Hc + h) * Wc + wgl;
    out[o] = acc0[r] + acc1[r] + bias2[ocr] + x[o];
  }
}

// ---------------------------------------------------------------------------
extern "C" void kernel_launch(void* const* d_in, const int* in_sizes, int n_in,
                              void* d_out, int out_size, void* d_ws, size_t ws_size,
                              hipStream_t stream) {
  const float* x    = (const float*)d_in[0];
  const float* wp_x = (const float*)d_in[1];
  const float* bp_x = (const float*)d_in[2];
  const float* wc_x = (const float*)d_in[3];
  const float* wm   = (const float*)d_in[4];
  const float* bm   = (const float*)d_in[5];
  const float* wp_y = (const float*)d_in[6];
  const float* bp_y = (const float*)d_in[7];
  const float* wc_y = (const float*)d_in[8];
  const float* wo   = (const float*)d_in[9];
  const float* bo   = (const float*)d_in[10];
  float* out = (float*)d_out;

  char* ws = (char*)d_ws;
  bf16_t* Aoff  = (bf16_t*)ws;                            // 64*576*2  =  73,728 B
  bf16_t* A     = (bf16_t*)(ws + 73728);                  // 64*1728*2 = 221,184 B
  float*  bias2 = (float*)(ws + 73728 + 221184);          // 256 B
  float*  bpoff = (float*)(ws + 73728 + 221184 + 256);    // 256 B

  pack_offw<<<(Cc * KOFF + 255) / 256, 256, 0, stream>>>(wp_x, wp_y, Aoff);
  fuse_weights<<<(Cc * KTOT + 255) / 256, 256, 0, stream>>>(wc_x, wm, wc_y, wo, A);
  fuse_bias<<<1, 128, 0, stream>>>(wo, bm, bo, bp_x, bp_y, bias2, bpoff);
  dfem_main<<<Bc * Hc * (Wc / 32), 256, 0, stream>>>(
      x, Aoff, A, bias2, bpoff, out);
}
